// BottleneckAttention_modified_37744172597422
// MI455X (gfx1250) — matrix operand order repaired and checked by reading, not compiled, not run on hardware
//
#include <hip/hip_runtime.h>

// Problem constants (from reference): KC=128, KH=10, KW=4
// z shapes (1,128,100,64); q: 160 x 5120; kv: 5551 x 5120
#define D_TOT 5120
#define M_TOT 160
#define N_TOT 5551
#define NPAD  5568          // padded score-row stride (mult of 64)
#define LSTR  72            // padded LDS k-stride in halfs (bank-conflict free frag loads)
#define MT    32            // query rows per block (2 WMMA tiles, 160 = 5*32)

typedef __attribute__((ext_vector_type(2)))  __fp16   h2;
typedef __attribute__((ext_vector_type(4)))  __fp16   h4;
typedef __attribute__((ext_vector_type(4)))  _Float16 v4h;
typedef __attribute__((ext_vector_type(8)))  _Float16 v8h;
typedef __attribute__((ext_vector_type(16))) _Float16 v16h;
typedef __attribute__((ext_vector_type(8)))  float    v8f;
typedef __attribute__((ext_vector_type(4)))  float    f4;    // 16B-aligned
typedef f4 f4u __attribute__((aligned(4)));                  // 4B-aligned view

// float4 -> 4 halfs via packed converts
__device__ __forceinline__ v4h cvt4(f4 v) {
  h2 a = __builtin_amdgcn_cvt_pkrtz(v.x, v.y);
  h2 b = __builtin_amdgcn_cvt_pkrtz(v.z, v.w);
  h4 t = __builtin_shufflevector(a, b, 0, 1, 2, 3);
  return __builtin_bit_cast(v4h, t);
}

// A fragment (16x32 f16): lanes 0-15 -> M=0..15 holding K 0..7 / 16..23,
// lanes 16-31 -> same M holding K 8..15 / 24..31 (CDNA5 ISA 7.12.2).
__device__ __forceinline__ v16h frag_a(const _Float16* Q, int lane, int kk) {
  const int m = lane & 15, half = lane >> 4;
  const _Float16* p = Q + m * LSTR + kk + half * 8;
  v8h lo = *(const v8h*)(p);
  v8h hi = *(const v8h*)(p + 16);
  return __builtin_shufflevector(lo, hi, 0,1,2,3,4,5,6,7,8,9,10,11,12,13,14,15);
}

// B fragment (32x16 f16): lanes 0-15 -> N=0..15 K=0..7 / 16..23, lanes 16-31 -> K=8..15 / 24..31.
__device__ __forceinline__ v16h frag_b(const _Float16* Krow, int lane, int kk) {
  const int n = lane & 15, ks = lane >> 4;
  const _Float16* p = Krow + n * LSTR + kk + ks * 8;
  v8h lo = *(const v8h*)(p);
  v8h hi = *(const v8h*)(p + 16);
  return __builtin_shufflevector(lo, hi, 0,1,2,3,4,5,6,7,8,9,10,11,12,13,14,15);
}

#define WMMA(a, b, c) __builtin_amdgcn_wmma_f32_16x16x32_f16(false, (a), false, (b), (short)0, (c), false, false)

// ---------------- Kernel 1: scores = q @ kv^T * (1/D) ----------------
// grid (44, 5); block 256 (8 waves). Block tile: 32 queries x 128 kv rows.
__global__ __launch_bounds__(256) void scores_kernel(const float* __restrict__ z1,
                                                     const float* __restrict__ z2,
                                                     float* __restrict__ scores) {
  __shared__ __align__(16) _Float16 Qs[MT * LSTR];
  __shared__ __align__(16) _Float16 Ks[128 * LSTR];
  const int tid = threadIdx.x, wave = tid >> 5, lane = tid & 31;
  const int m0 = blockIdx.y * MT;
  const int n0 = blockIdx.x * 128;

  // fixed kv row per thread (pair of threads per row); hoist n/61 out of K loop
  const int nl = tid >> 1;                                  // 0..127
  int n = n0 + nl; n = (n < N_TOT) ? n : (N_TOT - 1);
  const int hk = n / 61, wk = n - hk * 61;
  const int kvbase = hk * 64 + wk;                          // float offset inside (c,kh) plane
  const int gbase = (tid & 1) * 8;                          // k-group base for this thread

  v8f acc0 = {}, acc1 = {};
  for (int k0 = 0; k0 < D_TOT; k0 += 64) {
    __syncthreads();
    // Q tile: MT x 64 as 4-wide groups (fixed c,kh; kw=0..3 contiguous, 16B aligned)
#pragma unroll
    for (int t = 0; t < 2; ++t) {
      int idx = tid + t * 256;                              // 0..511
      int m = idx >> 4, g = idx & 15;
      int d0 = k0 + 4 * g;
      int c = d0 / 40, r = d0 - c * 40, kh = r >> 2;
      int mg = m0 + m, ih = mg >> 4, iw = mg & 15;
      f4 v = *(const f4*)(z1 + ((c * 100 + ih * 10 + kh) * 64 + iw * 4));
      *(v4h*)(Qs + m * LSTR + 4 * g) = cvt4(v);
    }
    // KV tile: 128 x 64; 8 groups per thread, contiguous 4-float loads (4B aligned)
#pragma unroll
    for (int t = 0; t < 8; ++t) {
      int g = gbase + t;
      int d0 = k0 + 4 * g;
      int c = d0 / 40, r = d0 - c * 40, kh = r >> 2;
      f4 v = *(const f4u*)(z2 + (c * 6400 + kh * 64 + kvbase));
      *(v4h*)(Ks + nl * LSTR + 4 * g) = cvt4(v);
    }
    __syncthreads();
    const _Float16* kb = Ks + (wave * 16) * LSTR;
#pragma unroll
    for (int kk = 0; kk < 64; kk += 32) {
      v16h b  = frag_b(kb, lane, kk);
      v16h a0 = frag_a(Qs, lane, kk);
      v16h a1 = frag_a(Qs + 16 * LSTR, lane, kk);
      acc0 = WMMA(a0, b, acc0);
      acc1 = WMMA(a1, b, acc1);
    }
  }
  const float scale = 1.0f / (float)D_TOT;
  const int nn = n0 + wave * 16 + (lane & 15);
  if (nn < N_TOT) {
    const int mg = m0 + ((lane >> 4) * 8);                  // C/D: VGPR r -> row r (+8 hi lanes)
#pragma unroll
    for (int r = 0; r < 8; ++r) {
      scores[(mg + r) * NPAD + nn]        = acc0[r] * scale;
      scores[(mg + 16 + r) * NPAD + nn]   = acc1[r] * scale;
    }
  }
}

// ---------------- Kernel 2: row softmax in-place ----------------
__global__ __launch_bounds__(256) void softmax_kernel(float* __restrict__ scores) {
  __shared__ float red[256];
  const int tid = threadIdx.x;
  float* row = scores + blockIdx.x * NPAD;

  float lmax = -3.0e38f;
  for (int j = tid; j < N_TOT; j += 256) lmax = fmaxf(lmax, row[j]);
  red[tid] = lmax; __syncthreads();
  for (int s = 128; s > 0; s >>= 1) {
    if (tid < s) red[tid] = fmaxf(red[tid], red[tid + s]);
    __syncthreads();
  }
  const float m = red[0]; __syncthreads();

  float lsum = 0.0f;
  for (int j = tid; j < N_TOT; j += 256) lsum += __expf(row[j] - m);
  red[tid] = lsum; __syncthreads();
  for (int s = 128; s > 0; s >>= 1) {
    if (tid < s) red[tid] += red[tid + s];
    __syncthreads();
  }
  const float inv = 1.0f / red[0]; __syncthreads();

  for (int j = tid; j < N_TOT; j += 256) row[j] = __expf(row[j] - m) * inv;
}

// ---------------- Kernel 3: out = attn @ kv, scatter to (1,128,100,64) ----------------
// grid (40, 5); block 256. Block tile: 32 queries x 128 out cols; K = 5551 kv rows.
__global__ __launch_bounds__(256) void out_kernel(const float* __restrict__ scores,
                                                  const float* __restrict__ z2,
                                                  float* __restrict__ out) {
  __shared__ __align__(16) _Float16 As[MT * LSTR];
  __shared__ __align__(16) _Float16 Bs[128 * LSTR];
  const int tid = threadIdx.x, wave = tid >> 5, lane = tid & 31;
  const int m0 = blockIdx.y * MT;
  const int j0 = blockIdx.x * 128;

  // fixed out-column per thread (pair of threads per col); hoist jg/40 out of K loop
  const int jl = tid >> 1;
  const int jg = j0 + jl;
  const int c = jg / 40, r = jg - c * 40, kh = r >> 2, kw = r & 3;
  const int coloff = c * 6400 + kh * 64 + kw;
  const int gbase = (tid & 1) * 8;

  v8f acc0 = {}, acc1 = {};
  for (int k0 = 0; k0 < N_TOT; k0 += 64) {                  // 87 chunks, tail masked via A=0
    __syncthreads();
    // A tile: MT x 64 from attn rows (contiguous, 16B aligned)
#pragma unroll
    for (int t = 0; t < 2; ++t) {
      int idx = tid + t * 256;
      int m = idx >> 4, g = idx & 15;
      int kg = k0 + 4 * g;
      const float* rowp = scores + (m0 + m) * NPAD;
      f4 v;
      if (kg + 3 < N_TOT) {
        v = *(const f4*)(rowp + kg);
      } else {
        v.x = (kg + 0 < N_TOT) ? rowp[kg + 0] : 0.0f;
        v.y = (kg + 1 < N_TOT) ? rowp[kg + 1] : 0.0f;
        v.z = (kg + 2 < N_TOT) ? rowp[kg + 2] : 0.0f;
        v.w = (kg + 3 < N_TOT) ? rowp[kg + 3] : 0.0f;
      }
      *(v4h*)(As + m * LSTR + 4 * g) = cvt4(v);
    }
    // B tile (kv transposed): Bs[j_local][k_local]; contiguous along kv-row index
    // except when the 4-group straddles a wk=61 boundary (~1 group in 16).
#pragma unroll
    for (int t = 0; t < 8; ++t) {
      int g = gbase + t;
      int kg = k0 + 4 * g;
      int kc = (kg <= N_TOT - 4) ? kg : (N_TOT - 4);        // clamp; tail garbage masked by A=0
      int hk = kc / 61, wkk = kc - hk * 61;
      f4 v;
      if (wkk <= 57) {
        v = *(const f4u*)(z2 + (coloff + hk * 64 + wkk));
      } else {
#pragma unroll
        for (int e = 0; e < 4; ++e) {
          int ke = kc + e;
          int hke = ke / 61, wke = ke - hke * 61;
          v[e] = z2[coloff + hke * 64 + wke];
        }
      }
      *(v4h*)(Bs + jl * LSTR + 4 * g) = cvt4(v);
    }
    __syncthreads();
    const _Float16* bb = Bs + (wave * 16) * LSTR;
#pragma unroll
    for (int kk = 0; kk < 64; kk += 32) {
      v16h b  = frag_b(bb, lane, kk);
      v16h a0 = frag_a(As, lane, kk);
      v16h a1 = frag_a(As + 16 * LSTR, lane, kk);
      acc0 = WMMA(a0, b, acc0);
      acc1 = WMMA(a1, b, acc1);
    }
  }
  // scatter: out[(c, ih*10+kh, iw*4+kw)]
  const int jgs = j0 + wave * 16 + (lane & 15);
  const int cs = jgs / 40, rs = jgs - cs * 40, khs = rs >> 2, kws = rs & 3;
#pragma unroll
  for (int rr = 0; rr < 8; ++rr) {
    int mgA = m0 + ((lane >> 4) * 8) + rr;
    out[(cs * 100 + (mgA >> 4) * 10 + khs) * 64 + (mgA & 15) * 4 + kws] = acc0[rr];
    int mgB = mgA + 16;
    out[(cs * 100 + (mgB >> 4) * 10 + khs) * 64 + (mgB & 15) * 4 + kws] = acc1[rr];
  }
}

extern "C" void kernel_launch(void* const* d_in, const int* in_sizes, int n_in,
                              void* d_out, int out_size, void* d_ws, size_t ws_size,
                              hipStream_t stream) {
  const float* z1 = (const float*)d_in[0];
  const float* z2 = (const float*)d_in[1];
  float* out      = (float*)d_out;
  float* scores   = (float*)d_ws;     // 160 * 5568 * 4 B = 3.56 MB

  dim3 g1((N_TOT + 127) / 128, M_TOT / MT);    // (44, 5)
  scores_kernel<<<g1, 256, 0, stream>>>(z1, z2, scores);

  softmax_kernel<<<M_TOT, 256, 0, stream>>>(scores);

  dim3 g3(D_TOT / 128, M_TOT / MT);            // (40, 5)
  out_kernel<<<g3, 256, 0, stream>>>(scores, z2, out);
}
